// GNNLoss_24481313587487
// MI455X (gfx1250) — compile-verified, matrix operand order checked by reading, not run on hardware
//
#include <hip/hip_runtime.h>
#include <hip/hip_bf16.h>

// ---------------------------------------------------------------------------
// CDNA5 / gfx1250 implementation.
// All heavy GEMMs use v_wmma_f32_16x16x32_f16 (wave32, 16x32 C-region per wave
// = 2 accumulators sharing one A fragment, K unrolled by 64 -> 4 back-to-back
// v_wmma per loop body, f32 accumulation).
// WMMA operands are kept natively f16 in memory: inputs/weights converted once,
// GEMM-feeding producers (scale/copy, l2norm, adjacency epilogue, gathers)
// write f16 directly. Adjacency matrices are exact in f16 (0/1 values,
// degrees <= 2048 = 2^11). Inner GEMM loop = contiguous b128 half loads + wmma,
// no conversions, no branches.
// Fragment layouts follow CDNA5 ISA 7.12.2:
//   A (16x32 f16): lane L (g=L>>4, r=L&15) holds row M=r,
//       element i -> K = k0 + ((i&8)?16:0) + (i&7) + 8*g
//   B: same K pattern, lane r = output column (B consumed "column major")
//   C (16x16 f32, 8 VGPRs): element r -> row = r + 8*g, col = lane&15
// ---------------------------------------------------------------------------

typedef __attribute__((ext_vector_type(16))) _Float16 v16h;
typedef __attribute__((ext_vector_type(8)))  float    v8f;

#define LANES 32

__device__ __forceinline__ float wave_sum(float v) {
#pragma unroll
  for (int off = 16; off > 0; off >>= 1) v += __shfl_xor(v, off, LANES);
  return v;
}
__device__ __forceinline__ float wave_max(float v) {
#pragma unroll
  for (int off = 16; off > 0; off >>= 1) v = fmaxf(v, __shfl_xor(v, off, LANES));
  return v;
}

// Contiguous fragment: two 8-half (16B) runs -> compiler emits global_load_b128.
__device__ __forceinline__ v16h frag_contig(const _Float16* __restrict__ p, int k0, int g) {
  v16h f;
#pragma unroll
  for (int i = 0; i < 16; ++i) {
    int kk = k0 + ((i & 8) << 1) + (i & 7) + (g << 3);
    f[i] = p[kk];
  }
  return f;
}
// Strided fragment (column gather of row-major B[K,N]).
__device__ __forceinline__ v16h frag_strided(const _Float16* __restrict__ p, long ld, int k0, int g) {
  v16h f;
#pragma unroll
  for (int i = 0; i < 16; ++i) {
    int kk = k0 + ((i & 8) << 1) + (i & 7) + (g << 3);
    f[i] = p[(long)kk * ld];
  }
  return f;
}

enum { EPI_PLAIN = 0, EPI_ADJ = 1 };

template <int EPI, typename CT>
__device__ __forceinline__ void store_tile(CT* __restrict__ C, int ldc, int mBase, int col,
                                           int g, v8f acc, const float* __restrict__ bias) {
#pragma unroll
  for (int i = 0; i < 8; ++i) {
    int row = mBase + i + (g << 3);
    float v = acc[i];
    if (EPI == EPI_PLAIN) {
      if (bias) v += bias[col];
    } else {                       // adjacency: (x > 0) ? 1 : 0, diag forced 1
      v = (v > 0.0f) ? 1.0f : 0.0f;
      if (row == col) v = 1.0f;
    }
    C[(size_t)row * ldc + col] = (CT)v;
  }
}

// C[M,N] = A[M,K] @ (TRANSB ? B[N,K]^T : B[K,N])   (+bias / adjacency epilogue)
// A,B are f16; C is f32 (PLAIN) or f16 (ADJ).
// One wave per 16x32 output region. M mult of 16, N mult of 32, K mult of 64.
template <int TRANSB, int EPI, typename CT>
__global__ void wmma_gemm_kernel(const _Float16* __restrict__ A, const _Float16* __restrict__ B,
                                 CT* __restrict__ C,
                                 int M, int N, int K, int lda, int ldb, int ldc,
                                 const float* __restrict__ bias) {
  const int lane  = threadIdx.x & 31;
  const int g     = lane >> 4;
  const int r     = lane & 15;
  const int nBase = blockIdx.x * 32;
  const int mBase = blockIdx.y * 16;

  const _Float16* pa = A + (size_t)(mBase + r) * lda;    // A row for this lane
  const _Float16* pb0;
  const _Float16* pb1;
  if (TRANSB) {                                          // B[N,K] row-major
    pb0 = B + (size_t)(nBase + r) * ldb;
    pb1 = pb0 + (size_t)16 * ldb;
  } else {                                               // B[K,N] row-major
    pb0 = B + (nBase + r);
    pb1 = pb0 + 16;
  }

  v8f acc0 = {};
  v8f acc1 = {};
  for (int k0 = 0; k0 < K; k0 += 64) {
    if (k0 + 128 <= K) {                 // next tiles -> global_prefetch_b8
      __builtin_prefetch(pa + k0 + 64, 0, 1);
      if (TRANSB) {
        __builtin_prefetch(pb0 + k0 + 64, 0, 1);
        __builtin_prefetch(pb1 + k0 + 64, 0, 1);
      }
    }
#pragma unroll
    for (int kh = 0; kh < 2; ++kh) {
      const int k = k0 + kh * 32;
      v16h a  = frag_contig(pa, k, g);
      v16h b0 = TRANSB ? frag_contig(pb0, k, g) : frag_strided(pb0, ldb, k, g);
      v16h b1 = TRANSB ? frag_contig(pb1, k, g) : frag_strided(pb1, ldb, k, g);
      acc0 = __builtin_amdgcn_wmma_f32_16x16x32_f16(false, a, false, b0,
                                                    (short)0, acc0, false, false);
      acc1 = __builtin_amdgcn_wmma_f32_16x16x32_f16(false, a, false, b1,
                                                    (short)0, acc1, false, false);
    }
  }

  store_tile<EPI, CT>(C, ldc, mBase, nBase + r,      g, acc0, bias);
  store_tile<EPI, CT>(C, ldc, mBase, nBase + r + 16, g, acc1, bias);
}

// Elementwise f32 -> f16 conversion (inputs / weights, once per buffer).
__global__ void cvt_f16_kernel(const float* __restrict__ src, _Float16* __restrict__ dst, int n) {
  int i = blockIdx.x * blockDim.x + threadIdx.x;
  if (i < n) dst[i] = (_Float16)src[i];
}

// Row-wise L2 normalize in place; optional f16 mirror for WMMA consumers.
__global__ void rows_l2norm_kernel(float* __restrict__ X, _Float16* __restrict__ Xh,
                                   int n, int d) {
  int lane = threadIdx.x;
  int row  = blockIdx.x * blockDim.y + threadIdx.y;
  if (row >= n) return;
  float* p = X + (size_t)row * d;
  float s = 0.0f;
  for (int c = lane; c < d; c += LANES) { float x = p[c]; s += x * x; }
  s = wave_sum(s);
  float inv = 1.0f / sqrtf(s);
  for (int c = lane; c < d; c += LANES) {
    float y = p[c] * inv;
    p[c] = y;
    if (Xh) Xh[(size_t)row * d + c] = (_Float16)y;
  }
}

// dinv[i] = rsqrt(max(rowsum(A_i), 1e-12)); f16 adjacency input, f32 accumulate.
__global__ void rows_sum_rsqrt_kernel(const _Float16* __restrict__ A, float* __restrict__ dinv,
                                      int n, int w) {
  int lane = threadIdx.x;
  int row  = blockIdx.x * blockDim.y + threadIdx.y;
  if (row >= n) return;
  const _Float16* p = A + (size_t)row * w;
  float s = 0.0f;
  for (int c = lane; c < w; c += LANES) s += (float)p[c];
  s = wave_sum(s);
  if (lane == 0) dinv[row] = rsqrtf(fmaxf(s, 1e-12f));
}

// dst_h[r*ldd + c] = (f16)(src[r*d + c] * s[r]^pw)   (s == nullptr -> copy)
__global__ void scale_copy_kernel(const float* __restrict__ src, const float* __restrict__ s,
                                  int pw, _Float16* __restrict__ dst, int n, int d, int ldd) {
  int i = blockIdx.x * blockDim.x + threadIdx.x;
  if (i >= n * d) return;
  int r = i / d, c = i % d;
  float v = src[(size_t)r * d + c];
  if (s) { float f = s[r]; if (pw == 2) f *= f; v *= f; }
  dst[(size_t)r * ldd + c] = (_Float16)v;
}

// out[i] = dot(X_i, Y_i); one wave per row (f32 features).
__global__ void rows_dot_kernel(const float* __restrict__ X, const float* __restrict__ Y,
                                float* __restrict__ out, int n, int d) {
  int lane = threadIdx.x;
  int row  = blockIdx.x * blockDim.y + threadIdx.y;
  if (row >= n) return;
  const float* px = X + (size_t)row * d;
  const float* py = Y + (size_t)row * d;
  float s = 0.0f;
  for (int c = lane; c < d; c += LANES) s += px[c] * py[c];
  s = wave_sum(s);
  if (lane == 0) out[row] = s;
}

// scores[i] = sigmoid(dot(F_i, w) + b0); one wave per row.
__global__ void scores_kernel(const float* __restrict__ F, const float* __restrict__ w,
                              const float* __restrict__ b, float* __restrict__ scores,
                              int n, int d) {
  int lane = threadIdx.x;
  int row  = blockIdx.x * blockDim.y + threadIdx.y;
  if (row >= n) return;
  const float* p = F + (size_t)row * d;
  float s = 0.0f;
  for (int c = lane; c < d; c += LANES) s += p[c] * w[c];
  s = wave_sum(s);
  if (lane == 0) {
    s += b[0];
    scores[row] = 1.0f / (1.0f + expf(-s));
  }
}

// Exact stable top-k via full bitonic sort of 2048 keys in LDS (desc, ties by idx).
__global__ void topk_kernel(const float* __restrict__ scores,
                            float* __restrict__ svals, int* __restrict__ sidx) {
  __shared__ float v[2048];
  __shared__ int   id[2048];
  int t = threadIdx.x;                       // blockDim.x == 1024
  v[t] = scores[t];               id[t] = t;
  v[t + 1024] = scores[t + 1024]; id[t + 1024] = t + 1024;
  __syncthreads();
  for (int k = 2; k <= 2048; k <<= 1) {
    for (int j = k >> 1; j > 0; j >>= 1) {
#pragma unroll
      for (int base = 0; base < 2048; base += 1024) {
        int i   = base + t;
        int ixj = i ^ j;
        if (ixj > i) {
          bool desc = ((i & k) == 0);
          bool bad  = (v[i] < v[ixj]) || (v[i] == v[ixj] && id[i] > id[ixj]);
          if (desc ? bad : !bad) {
            float tv = v[i]; v[i] = v[ixj]; v[ixj] = tv;
            int   ti = id[i]; id[i] = id[ixj]; id[ixj] = ti;
          }
        }
      }
      __syncthreads();
    }
  }
  svals[t] = v[t];
  sidx[t]  = id[t];
}

// dst[r,:] = src[idx[r],:] * vals[r]   (f32 features)
__global__ void gather_scale_kernel(const float* __restrict__ src, const int* __restrict__ idx,
                                    const float* __restrict__ vals, float* __restrict__ dst,
                                    int k, int d) {
  int i = blockIdx.x * blockDim.x + threadIdx.x;
  if (i >= k * d) return;
  int r = i / d, c = i % d;
  dst[i] = src[(size_t)idx[r] * d + c] * vals[r];
}

// dst[r,:] = A[idx[r],:]   (f16 adjacency rows)
__global__ void gather_rows_kernel(const _Float16* __restrict__ A, const int* __restrict__ idx,
                                   _Float16* __restrict__ dst, int k, int w) {
  int i = blockIdx.x * blockDim.x + threadIdx.x;
  if (i >= k * w) return;
  int r = i / w, c = i % w;
  dst[i] = A[(size_t)idx[r] * w + c];
}

// Per-row NCE loss term: logsumexp over {pos/T, -10/T (diag), G_row_or_col/T} - pos/T.
__global__ void nce_loss_kernel(const float* __restrict__ G, const float* __restrict__ pos,
                                float* __restrict__ lrow, int n, int colScan) {
  int lane = threadIdx.x;
  int row  = blockIdx.x * blockDim.y + threadIdx.y;
  if (row >= n) return;
  const float invT = 1.0f / 0.07f;
  float zpos = pos[row] * invT;
  float zd   = -10.0f * invT;
  float m = fmaxf(zpos, zd);
  for (int j = lane; j < n; j += LANES)
    if (j != row) {
      float z = (colScan ? G[(size_t)j * n + row] : G[(size_t)row * n + j]) * invT;
      m = fmaxf(m, z);
    }
  m = wave_max(m);
  float ssum = 0.0f;
  for (int j = lane; j < n; j += LANES)
    if (j != row) {
      float z = (colScan ? G[(size_t)j * n + row] : G[(size_t)row * n + j]) * invT;
      ssum += expf(z - m);
    }
  ssum = wave_sum(ssum);
  if (lane == 0) {
    ssum += expf(zpos - m) + expf(zd - m);
    lrow[row] = logf(ssum) + m - zpos;
  }
}

// Deterministic final reduction: mean over each NCE direction, summed.
__global__ void final_reduce_kernel(const float* __restrict__ lgs, const float* __restrict__ lgt,
                                    const float* __restrict__ lps, const float* __restrict__ lpt,
                                    float* __restrict__ out) {
  __shared__ float sm[256];
  int t = threadIdx.x;
  float a = 0.0f;
  for (int i = t; i < 2048; i += 256) a += lgs[i] + lgt[i];
  float b = 0.0f;
  for (int i = t; i < 1024; i += 256) b += lps[i] + lpt[i];
  sm[t] = a * (1.0f / 2048.0f) + b * (1.0f / 1024.0f);
  __syncthreads();
  for (int s = 128; s > 0; s >>= 1) {
    if (t < s) sm[t] += sm[t + s];
    __syncthreads();
  }
  if (t == 0) out[0] = sm[0];
}

extern "C" void kernel_launch(void* const* d_in, const int* in_sizes, int n_in,
                              void* d_out, int out_size, void* d_ws, size_t ws_size,
                              hipStream_t stream) {
  (void)in_sizes; (void)n_in; (void)out_size; (void)ws_size;
  constexpr int N = 2048, C = 512, D = 128, KP = 1024;

  const float* fs      = (const float*)d_in[0];
  const float* ft      = (const float*)d_in[1];
  const float* W_embed = (const float*)d_in[2];
  const float* b_embed = (const float*)d_in[3];
  const float* W_gnn   = (const float*)d_in[4];
  const float* b_gnn   = (const float*)d_in[5];
  const float* W_pool  = (const float*)d_in[6];
  const float* b_pool  = (const float*)d_in[7];
  const float* W_gnnp  = (const float*)d_in[8];
  const float* b_gnnp  = (const float*)d_in[9];

  // ---- workspace layout (~55 MB; fits entirely in the 192 MB L2) ----
  float* w = (float*)d_ws;
  size_t o = 0;
  auto allocf = [&](size_t cnt) { float* p = w + o; o += cnt; return p; };
  auto alloch = [&](size_t cnt) { _Float16* p = (_Float16*)(w + o); o += (cnt + 1) / 2; return p; };

  _Float16* fs_h   = alloch((size_t)N * C);
  _Float16* ft_h   = alloch((size_t)N * C);
  _Float16* We_h   = alloch((size_t)C * D);
  _Float16* Wg_h   = alloch((size_t)3 * D * D);
  _Float16* Wgp_h  = alloch((size_t)2 * D * D);
  float*    f_es   = allocf((size_t)N * D);
  float*    f_et   = allocf((size_t)N * D);
  _Float16* f_et_h = alloch((size_t)N * D);
  _Float16* Adj_h  = alloch((size_t)N * N);
  float*    dinv   = allocf(N);
  _Float16* tmp_h  = alloch((size_t)N * D);
  float*    u      = allocf((size_t)N * D);
  _Float16* cat_s  = alloch((size_t)N * 3 * D);
  _Float16* cat_t  = alloch((size_t)N * 3 * D);
  float*    f_gs   = allocf((size_t)N * D);
  float*    f_gt   = allocf((size_t)N * D);
  _Float16* f_gs_h = alloch((size_t)N * D);
  _Float16* f_gt_h = alloch((size_t)N * D);
  float*    G      = allocf((size_t)N * N);
  float*    pos    = allocf(N);
  float*    lg_s   = allocf(N);
  float*    lg_t   = allocf(N);
  float*    scr    = allocf(N);
  float*    svals  = allocf(N);
  int*      sidx   = (int*)allocf(N);
  float*    new_hs = allocf((size_t)KP * D);
  float*    new_ht = allocf((size_t)KP * D);
  _Float16* Ar_h   = alloch((size_t)KP * N);
  _Float16* Asub_h = alloch((size_t)KP * KP);
  float*    dinv2  = allocf(KP);
  _Float16* tmp2_h = alloch((size_t)KP * D);
  float*    u2     = allocf((size_t)KP * D);
  _Float16* cat2_s = alloch((size_t)KP * 2 * D);
  _Float16* cat2_t = alloch((size_t)KP * 2 * D);
  float*    f_ps   = allocf((size_t)KP * D);
  float*    f_pt   = allocf((size_t)KP * D);
  _Float16* f_ps_h = alloch((size_t)KP * D);
  _Float16* f_pt_h = alloch((size_t)KP * D);
  float*    G2     = allocf((size_t)KP * KP);
  float*    pos2   = allocf(KP);
  float*    lp_s   = allocf(KP);
  float*    lp_t   = allocf(KP);

  // ---- launch helpers ----
  auto gemm = [&](const _Float16* A, const _Float16* B, float* Cp, int M, int Nn, int K,
                  int lda, int ldb, int ldc, const float* bias, int transB) {
    dim3 grid(Nn / 32, M / 16);
    if (transB)
      wmma_gemm_kernel<1, EPI_PLAIN, float><<<grid, 32, 0, stream>>>(A, B, Cp, M, Nn, K,
                                                                     lda, ldb, ldc, bias);
    else
      wmma_gemm_kernel<0, EPI_PLAIN, float><<<grid, 32, 0, stream>>>(A, B, Cp, M, Nn, K,
                                                                     lda, ldb, ldc, bias);
  };
  auto gemm_adj = [&](const _Float16* A, const _Float16* B, _Float16* Cp, int M, int Nn, int K,
                      int lda, int ldb, int ldc) {
    dim3 grid(Nn / 32, M / 16);
    wmma_gemm_kernel<1, EPI_ADJ, _Float16><<<grid, 32, 0, stream>>>(A, B, Cp, M, Nn, K,
                                                                    lda, ldb, ldc, nullptr);
  };
  auto cvt = [&](const float* src, _Float16* dst, int n) {
    cvt_f16_kernel<<<(n + 255) / 256, 256, 0, stream>>>(src, dst, n);
  };
  auto l2norm = [&](float* X, _Float16* Xh, int n, int d) {
    dim3 blk(32, 8);
    rows_l2norm_kernel<<<n / 8, blk, 0, stream>>>(X, Xh, n, d);
  };
  auto scale_copy = [&](const float* src, const float* s, int pw, _Float16* dst,
                        int n, int d, int ldd) {
    int total = n * d;
    scale_copy_kernel<<<(total + 255) / 256, 256, 0, stream>>>(src, s, pw, dst, n, d, ldd);
  };

  // ---- 0) one-time f16 conversion of inputs & weights ----
  cvt(fs, fs_h, N * C);
  cvt(ft, ft_h, N * C);
  cvt(W_embed, We_h, C * D);
  cvt(W_gnn, Wg_h, 3 * D * D);
  cvt(W_gnnp, Wgp_h, 2 * D * D);

  // ---- 1) embed + l2norm ----
  gemm(fs_h, We_h, f_es, N, D, C, C, D, D, b_embed, 0);
  l2norm(f_es, nullptr, N, D);
  gemm(ft_h, We_h, f_et, N, D, C, C, D, D, b_embed, 0);
  l2norm(f_et, f_et_h, N, D);

  // ---- 2) adjacency A = (f_et f_et^T > 0), diag = 1 (exact in f16); degrees ----
  gemm_adj(f_et_h, f_et_h, Adj_h, N, N, D, D, D, N);
  {
    dim3 blk(32, 8);
    rows_sum_rsqrt_kernel<<<N / 8, blk, 0, stream>>>(Adj_h, dinv, N, N);
  }

  // ---- 3) TAGConv k=2 per branch: cat=[h | d(A dh) | d(A d^2(A dh))] @ W + b ----
  auto tag = [&](const float* fe, _Float16* cat, float* fg, _Float16* fgh) {
    scale_copy(fe, nullptr, 1, cat, N, D, 3 * D);        // hop 0
    scale_copy(fe, dinv, 1, tmp_h, N, D, D);             // dinv * h
    gemm(Adj_h, tmp_h, u, N, D, N, N, D, D, nullptr, 0);
    scale_copy(u, dinv, 1, cat + D, N, D, 3 * D);        // hop 1
    scale_copy(u, dinv, 2, tmp_h, N, D, D);              // dinv^2 * u -> next input
    gemm(Adj_h, tmp_h, u, N, D, N, N, D, D, nullptr, 0);
    scale_copy(u, dinv, 1, cat + 2 * D, N, D, 3 * D);    // hop 2
    gemm(cat, Wg_h, fg, N, D, 3 * D, 3 * D, D, D, b_gnn, 0);
    l2norm(fg, fgh, N, D);
  };
  tag(f_et, cat_t, f_gt, f_gt_h);
  tag(f_es, cat_s, f_gs, f_gs_h);

  // ---- 4) graph-level NCE: one gram serves both directions ----
  {
    dim3 blk(32, 8);
    rows_dot_kernel<<<N / 8, blk, 0, stream>>>(f_gt, f_gs, pos, N, D);
  }
  gemm(f_gt_h, f_gs_h, G, N, N, D, D, D, N, nullptr, 1);
  {
    dim3 blk(32, 8);
    nce_loss_kernel<<<N / 8, blk, 0, stream>>>(G, pos, lg_s, N, 0); // rows  (loss_s)
    nce_loss_kernel<<<N / 8, blk, 0, stream>>>(G, pos, lg_t, N, 1); // cols  (loss_t)
  }

  // ---- 5) pooling: scores, exact top-k, gather ----
  {
    dim3 blk(32, 8);
    scores_kernel<<<N / 8, blk, 0, stream>>>(f_gt, W_pool, b_pool, scr, N, D);
  }
  topk_kernel<<<1, 1024, 0, stream>>>(scr, svals, sidx);
  gather_scale_kernel<<<(KP * D + 255) / 256, 256, 0, stream>>>(f_gs, sidx, svals, new_hs, KP, D);
  gather_scale_kernel<<<(KP * D + 255) / 256, 256, 0, stream>>>(f_gt, sidx, svals, new_ht, KP, D);

  // sub = (A[idx,:] @ A[idx,:]^T > 0), diag = 1   (A symmetric -> gram form)
  gather_rows_kernel<<<((size_t)KP * N + 255) / 256, 256, 0, stream>>>(Adj_h, sidx, Ar_h, KP, N);
  gemm_adj(Ar_h, Ar_h, Asub_h, KP, KP, N, N, N, KP);
  {
    dim3 blk(32, 8);
    rows_sum_rsqrt_kernel<<<KP / 8, blk, 0, stream>>>(Asub_h, dinv2, KP, KP);
  }

  // ---- 6) pooled TAGConv k=1 per branch ----
  auto tagp = [&](const float* h, _Float16* cat2, float* fp, _Float16* fph) {
    scale_copy(h, nullptr, 1, cat2, KP, D, 2 * D);
    scale_copy(h, dinv2, 1, tmp2_h, KP, D, D);
    gemm(Asub_h, tmp2_h, u2, KP, D, KP, KP, D, D, nullptr, 0);
    scale_copy(u2, dinv2, 1, cat2 + D, KP, D, 2 * D);
    gemm(cat2, Wgp_h, fp, KP, D, 2 * D, 2 * D, D, D, b_gnnp, 0);
    l2norm(fp, fph, KP, D);
  };
  tagp(new_ht, cat2_t, f_pt, f_pt_h);
  tagp(new_hs, cat2_s, f_ps, f_ps_h);

  // ---- 7) pooled NCE ----
  {
    dim3 blk(32, 8);
    rows_dot_kernel<<<KP / 8, blk, 0, stream>>>(f_pt, f_ps, pos2, KP, D);
  }
  gemm(f_pt_h, f_ps_h, G2, KP, KP, D, D, D, KP, nullptr, 1);
  {
    dim3 blk(32, 8);
    nce_loss_kernel<<<KP / 8, blk, 0, stream>>>(G2, pos2, lp_s, KP, 0);
    nce_loss_kernel<<<KP / 8, blk, 0, stream>>>(G2, pos2, lp_t, KP, 1);
  }

  // ---- 8) deterministic scalar reduction ----
  final_reduce_kernel<<<1, 256, 0, stream>>>(lg_s, lg_t, lp_s, lp_t, (float*)d_out);
}
